// GCNhlayer_12635793785486
// MI455X (gfx1250) — matrix operand order repaired and checked by reading, not compiled, not run on hardware
//
#include <hip/hip_runtime.h>
#include <hip/hip_bf16.h>

typedef __attribute__((ext_vector_type(16))) _Float16 v16h;
typedef __attribute__((ext_vector_type(8)))  float    v8f;

#define N_NODES 100000
#define N_EDGES 1600000
#define DIM     128
#define LN_EPS  1e-5f

// Hardware float atomic add (global_atomic_add_f32, no-return form).
__device__ __forceinline__ void atomic_add_f32(float* p, float v) {
    __hip_atomic_fetch_add(p, v, __ATOMIC_RELAXED, __HIP_MEMORY_SCOPE_AGENT);
}

// ---------------------------------------------------------------------------
// 1) init: zero the aggregation buffer (d_out) with b128 stores and set
//    deg[n]=1 (self loop).  grid = N*DIM/(256*4) = 12500 blocks, exact.
__global__ __launch_bounds__(256) void k_init(float* __restrict__ agg,
                                              float* __restrict__ deg) {
    size_t i = (size_t)blockIdx.x * 256 + threadIdx.x;
    float4 z = {0.0f, 0.0f, 0.0f, 0.0f};
    *(float4*)(agg + i * 4) = z;
    if (i < N_NODES) deg[i] = 1.0f;
}

// ---------------------------------------------------------------------------
// 2) degree: atomic +1 per edge destination.  grid = E/256 = 6250, exact
__global__ __launch_bounds__(256) void k_degree(const long long* __restrict__ dst,
                                                float* __restrict__ deg) {
    int e = blockIdx.x * 256 + threadIdx.x;
    atomic_add_f32(&deg[(int)dst[e]], 1.0f);
}

// ---------------------------------------------------------------------------
// 3) dinv = rsqrt(deg)  (deg >= 1 always, thanks to self loops)
__global__ __launch_bounds__(256) void k_rsqrt(float* __restrict__ deg) {
    int i = blockIdx.x * 256 + threadIdx.x;
    if (i < N_NODES) deg[i] = rsqrtf(deg[i]);
}

// ---------------------------------------------------------------------------
// 4) pack W (DIMxDIM f32, row-major K x N) into WMMA B-operand layout as
//    f16 hi/lo split.  B in this code's contiguous k order: VGPR v holds K=2v,2v+1 for lanes
//    0-15 (N=lane) and K=16+2v,16+2v+1 for lanes 16-31 (N=lane-16)
//    => element i of v16h is simply K = kblk*32 + (lane>=16?16:0) + i.
//    Packed index: (((kb*8 + ct)*32 + lane)*16 + i) == t*16 + i.
__global__ __launch_bounds__(256) void k_packw(const float* __restrict__ W,
                                               _Float16* __restrict__ Bh,
                                               _Float16* __restrict__ Bl) {
    int t    = blockIdx.x * 256 + threadIdx.x;   // 4 blocks, 1024 threads
    int kb   = t >> 8;
    int ct   = (t >> 5) & 7;
    int lane = t & 31;
    int k0   = kb * 32 + (lane >> 4) * 16;
    int n    = ct * 16 + (lane & 15);
    int base = t << 4;
#pragma unroll
    for (int i = 0; i < 16; ++i) {
        float    w  = W[(k0 + i) * DIM + n];
        _Float16 hi = (_Float16)w;
        Bh[base + i] = hi;
        Bl[base + i] = (_Float16)(w - (float)hi);
    }
}

// ---------------------------------------------------------------------------
// 5) GEMM h = x @ W with split-f16 WMMA (fp32-accurate):
//    h = xh*Wh + xh*Wl + xl*Wh, fp32 accumulate (error ~2^-22).
//    One 16x16 output tile per wave; block = (32,8): 8 col tiles of one
//    16-row stripe.  grid = N/16 = 6250 blocks, exact.
__global__ __launch_bounds__(256) void k_gemm(const float* __restrict__ x,
                                              const _Float16* __restrict__ Bh,
                                              const _Float16* __restrict__ Bl,
                                              float* __restrict__ h) {
    const int lane = threadIdx.x;            // 0..31
    const int ct   = threadIdx.y;            // 0..7  (col tile)
    const int row0 = blockIdx.x * 16;
    const int m    = lane & 15;              // A-matrix row within tile
    const int koff = (lane >> 4) * 16;       // lanes 16-31 hold K+16 halves

    const float* xrow = x + (size_t)(row0 + m) * DIM;
    v8f c = {};

#pragma unroll
    for (int kb = 0; kb < 4; ++kb) {
        // A in this code's contiguous k order: VGPR0-3 -> K = kb*32 + koff + 0..7
        //                     VGPR4-7 -> K = kb*32 + koff + 8..15
        float af[16];
        const float* p0 = xrow + kb * 32 + koff;
        const float* p1 = p0 + 8;
#pragma unroll
        for (int i = 0; i < 8; ++i) { af[i] = p0[i]; af[8 + i] = p1[i]; }

        v16h ah, al;
#pragma unroll
        for (int i = 0; i < 16; ++i) {
            _Float16 hi = (_Float16)af[i];
            ah[i] = hi;
            al[i] = (_Float16)(af[i] - (float)hi);
        }

        const v16h bh = *(const v16h*)(Bh + (((kb * 8 + ct) * 32 + lane) << 4));
        const v16h bl = *(const v16h*)(Bl + (((kb * 8 + ct) * 32 + lane) << 4));

        c = __builtin_amdgcn_wmma_f32_16x16x32_f16(false, ah, false, bh,
                                                   (short)0, c, false, false);
        c = __builtin_amdgcn_wmma_f32_16x16x32_f16(false, ah, false, bl,
                                                   (short)0, c, false, false);
        c = __builtin_amdgcn_wmma_f32_16x16x32_f16(false, al, false, bh,
                                                   (short)0, c, false, false);
    }

    // C/D layout: VGPR r -> row M=r (lanes 0-15, N=lane) / M=8+r (lanes 16-31)
    float* hout = h + (size_t)(row0 + (lane >> 4) * 8) * DIM + ct * 16 + (lane & 15);
#pragma unroll
    for (int r = 0; r < 8; ++r) hout[(size_t)r * DIM] = c[r];
}

// ---------------------------------------------------------------------------
// 6) edge scatter: one wave per edge, float4 per lane (32*4 = 128 floats).
//    h (51MB) and agg (51MB) both live in the 192MB L2, so the gather and
//    the atomic RMWs are L2-resident.  grid = E/8 = 200000 blocks, exact.
__global__ __launch_bounds__(256) void k_scatter(const long long* __restrict__ src,
                                                 const long long* __restrict__ dst,
                                                 const float* __restrict__ dinv,
                                                 const float* __restrict__ h,
                                                 float* __restrict__ agg) {
    const int lane = threadIdx.x;
    const int e    = blockIdx.x * 8 + threadIdx.y;
    const int s    = (int)src[e];
    const int d    = (int)dst[e];
    const float en = dinv[s] * dinv[d];

    const float4 v = *(const float4*)(h + (size_t)s * DIM + lane * 4);
    float* o = agg + (size_t)d * DIM + lane * 4;
    atomic_add_f32(o + 0, v.x * en);
    atomic_add_f32(o + 1, v.y * en);
    atomic_add_f32(o + 2, v.z * en);
    atomic_add_f32(o + 3, v.w * en);
}

// ---------------------------------------------------------------------------
// 7) LayerNorm + self-loop fold + bias + ReLU.  Wave per row, shfl_xor
//    reduction (wave32).  grid = N/8 = 12500 blocks of (32,8), exact.
__global__ __launch_bounds__(256) void k_layernorm(const float* __restrict__ h,
                                                   const float* __restrict__ dinv,
                                                   const float* __restrict__ bias,
                                                   const float* __restrict__ gamma,
                                                   const float* __restrict__ beta,
                                                   float* __restrict__ out) {
    const int lane = threadIdx.x;
    const int row  = blockIdx.x * 8 + threadIdx.y;

    const float di  = dinv[row];
    const float dl2 = di * di;                   // self-loop enorm

    const float4 a  = *(const float4*)(out  + (size_t)row * DIM + lane * 4);
    const float4 hv = *(const float4*)(h    + (size_t)row * DIM + lane * 4);
    const float4 bb = *(const float4*)(bias + lane * 4);

    float v0 = a.x + hv.x * dl2 + bb.x;
    float v1 = a.y + hv.y * dl2 + bb.y;
    float v2 = a.z + hv.z * dl2 + bb.z;
    float v3 = a.w + hv.w * dl2 + bb.w;

    float sum = v0 + v1 + v2 + v3;
    float sq  = v0 * v0 + v1 * v1 + v2 * v2 + v3 * v3;
#pragma unroll
    for (int off = 16; off > 0; off >>= 1) {
        sum += __shfl_xor(sum, off, 32);
        sq  += __shfl_xor(sq,  off, 32);
    }
    const float mu  = sum * (1.0f / DIM);
    const float var = sq * (1.0f / DIM) - mu * mu;
    const float inv = rsqrtf(var + LN_EPS);

    const float4 g  = *(const float4*)(gamma + lane * 4);
    const float4 be = *(const float4*)(beta  + lane * 4);

    float4 y;
    y.x = fmaxf((v0 - mu) * inv * g.x + be.x, 0.0f);
    y.y = fmaxf((v1 - mu) * inv * g.y + be.y, 0.0f);
    y.z = fmaxf((v2 - mu) * inv * g.z + be.z, 0.0f);
    y.w = fmaxf((v3 - mu) * inv * g.w + be.w, 0.0f);
    *(float4*)(out + (size_t)row * DIM + lane * 4) = y;
}

// ---------------------------------------------------------------------------
extern "C" void kernel_launch(void* const* d_in, const int* in_sizes, int n_in,
                              void* d_out, int out_size, void* d_ws, size_t ws_size,
                              hipStream_t stream) {
    const float*     x     = (const float*)d_in[0];
    const long long* ei    = (const long long*)d_in[1];   // [2, E] int64
    const float*     W     = (const float*)d_in[2];
    const float*     bias  = (const float*)d_in[3];
    const float*     gamma = (const float*)d_in[4];
    const float*     beta  = (const float*)d_in[5];
    float*           out   = (float*)d_out;

    const long long* src = ei;
    const long long* dst = ei + N_EDGES;

    // workspace layout
    char*     ws   = (char*)d_ws;
    float*    dinv = (float*)ws;                                   // N floats (deg -> rsqrt in place)
    float*    h    = (float*)(ws + (512ull << 10));                // N*DIM floats
    _Float16* Bh   = (_Float16*)(ws + (512ull << 10) + (size_t)N_NODES * DIM * 4);
    _Float16* Bl   = Bh + 4 * 8 * 32 * 16;                         // 16384 halves each

    dim3 w8(32, 8);

    k_init     <<<(N_NODES * DIM) / 1024, 256, 0, stream>>>(out, dinv);
    k_degree   <<<N_EDGES / 256,          256, 0, stream>>>(dst, dinv);
    k_rsqrt    <<<(N_NODES + 255) / 256,  256, 0, stream>>>(dinv);
    k_packw    <<<4,                      256, 0, stream>>>(W, Bh, Bl);
    k_gemm     <<<N_NODES / 16,           w8,  0, stream>>>(x, Bh, Bl, h);
    k_scatter  <<<N_EDGES / 8,            w8,  0, stream>>>(src, dst, dinv, h, out);
    k_layernorm<<<N_NODES / 8,            w8,  0, stream>>>(h, dinv, bias, gamma, beta, out);
}
